// ROI_2156073583248
// MI455X (gfx1250) — matrix operand order repaired and checked by reading, not compiled, not run on hardware
//
#include <hip/hip_runtime.h>
#include <hip/hip_bf16.h>
#include <math.h>

typedef __attribute__((ext_vector_type(8)))  _Float16 v8h;
typedef __attribute__((ext_vector_type(16))) _Float16 v16h;
typedef __attribute__((ext_vector_type(8)))  float    v8f;

// Problem constants (fixed by setup_inputs)
#define RN        512          // rois
#define CN        512          // channels
#define HF        50
#define WF        50
#define POOLN     7
#define KBINN     9
#define DN        (CN * POOLN * POOLN)   // 25088
#define HID       1024
#define NCLS      21
#define NFG       20
#define NCAND     (RN * NFG)             // 10240
#define NPAD      16384
#define K2        2048
#define NOUT      100
#define BBOX_CLIP 4.135166556742356f     // log(1000/16)
#define SCORE_THv 0.05f
#define NMS_THv   0.3f
#define MIN_SZ    16.0f

// ---------------------------------------------------------------------------
// Transpose + fp32->f16 convert: src [K,N] row-major -> dst [N,K] row-major
// ---------------------------------------------------------------------------
__global__ __launch_bounds__(256) void convt_kernel(const float* __restrict__ src,
                                                    _Float16* __restrict__ dst,
                                                    int K, int N) {
  __shared__ float tile[32][33];
  const int k0 = blockIdx.x * 32;
  const int n0 = blockIdx.y * 32;
  const int tx = threadIdx.x;   // 0..31
  const int ty = threadIdx.y;   // 0..7
#pragma unroll
  for (int r = ty; r < 32; r += 8)
    tile[r][tx] = src[(size_t)(k0 + r) * N + (n0 + tx)];
  __syncthreads();
#pragma unroll
  for (int r = ty; r < 32; r += 8)
    dst[(size_t)(n0 + r) * K + (k0 + tx)] = (_Float16)tile[tx][r];
}

// ---------------------------------------------------------------------------
// ROI max pool (legacy Fast R-CNN semantics) -> f16 activation matrix [R, D]
// ---------------------------------------------------------------------------
__global__ __launch_bounds__(256) void roi_pool_kernel(const float* __restrict__ feat,
                                                       const float* __restrict__ rois,
                                                       const int* __restrict__ imgw_p,
                                                       _Float16* __restrict__ poolA) {
  const int r = blockIdx.x;
  const float scale = exp2f(rintf(log2f((float)WF / (float)(*imgw_p))));
  const int x1 = (int)rintf(rois[r * 4 + 0] * scale);
  const int y1 = (int)rintf(rois[r * 4 + 1] * scale);
  const int x2 = (int)rintf(rois[r * 4 + 2] * scale);
  const int y2 = (int)rintf(rois[r * 4 + 3] * scale);
  const float roi_w = (float)max(x2 - x1 + 1, 1);
  const float roi_h = (float)max(y2 - y1 + 1, 1);
  const float bw = roi_w / (float)POOLN;
  const float bh = roi_h / (float)POOLN;

  for (int t = threadIdx.x; t < CN * POOLN * POOLN; t += blockDim.x) {
    const int c = t / (POOLN * POOLN);
    const int bin = t % (POOLN * POOLN);
    const int ph = bin / POOLN;
    const int pw = bin % POOLN;
    int hs = (int)floorf((float)ph * bh) + y1;
    int he = (int)ceilf((float)(ph + 1) * bh) + y1;
    int ws = (int)floorf((float)pw * bw) + x1;
    int we = (int)ceilf((float)(pw + 1) * bw) + x1;
    hs = min(max(hs, 0), HF); he = min(max(he, 0), HF);
    ws = min(max(ws, 0), WF); we = min(max(we, 0), WF);
    const float* fc = feat + (size_t)c * HF * WF;
    float v = -INFINITY;
    bool any = false;
#pragma unroll
    for (int kh = 0; kh < KBINN; ++kh) {
      const int hi = hs + kh;
      if (hi >= he) continue;
      const int hic = min(hi, HF - 1);
#pragma unroll
      for (int kw = 0; kw < KBINN; ++kw) {
        const int wi = ws + kw;
        if (wi >= we) continue;
        const int wic = min(wi, WF - 1);
        v = fmaxf(v, fc[hic * WF + wic]);
        any = true;
      }
    }
    poolA[(size_t)r * DN + c * (POOLN * POOLN) + ph * POOLN + pw] =
        (_Float16)(any ? v : 0.0f);
  }
}

// ---------------------------------------------------------------------------
// Fragment load helpers (CDNA5 ISA 7.12.2 wave32 layouts)
// A frag (16x32 f16): lanes 0-15 row m, elems {K0..7, K16..23};
//                     lanes 16-31 row m, elems {K8..15, K24..31}
// B frag (32x16 f16): lanes 0-15 col n, elems {K0..7, K16..23}; lanes 16-31 {K8..15, K24..31}
// ---------------------------------------------------------------------------
__device__ __forceinline__ v16h load_afrag(const _Float16* arow, int k0, bool hiHalf) {
  const _Float16* ap = arow + k0 + (hiHalf ? 8 : 0);
  v8h lo = *(const v8h*)(ap);
  v8h hi = *(const v8h*)(ap + 16);
  return __builtin_shufflevector(lo, hi, 0, 1, 2, 3, 4, 5, 6, 7,
                                 8, 9, 10, 11, 12, 13, 14, 15);
}
__device__ __forceinline__ v16h load_bfrag(const _Float16* brow, int k0, bool hiHalf) {
  const _Float16* bp = brow + k0 + (hiHalf ? 8 : 0);
  v8h lo = *(const v8h*)(bp);
  v8h hi = *(const v8h*)(bp + 16);
  return __builtin_shufflevector(lo, hi, 0, 1, 2, 3, 4, 5, 6, 7,
                                 8, 9, 10, 11, 12, 13, 14, 15);
}

// ---------------------------------------------------------------------------
// WMMA f16 GEMM:  Out[M,N] = relu(A[M,K] * Bt[N,K]^T + bias)  (f16 out)
// Wave tile 32x64 (8x v_wmma_f32_16x16x32_f16 per K-step, A-frags reused 4x,
// B-frags reused 2x). 4 waves/block (2m x 2n) -> block tile 64x128.
// Software-pipelined: next K-step fragments are loaded into registers while
// the current K-step's 8 WMMAs execute, so waits land behind matrix work.
// ---------------------------------------------------------------------------
__global__ __launch_bounds__(128) void gemm_wmma_kernel(const _Float16* __restrict__ A,
                                                        const _Float16* __restrict__ Bt,
                                                        const float* __restrict__ bias,
                                                        _Float16* __restrict__ Out,
                                                        int M, int N, int K) {
  const int lane = threadIdx.x & 31;
  const int wave = threadIdx.x >> 5;      // 0..3
  const int m0 = blockIdx.x * 64 + (wave & 1) * 32;
  const int n0 = blockIdx.y * 128 + (wave >> 1) * 64;
  const int l15 = lane & 15;
  const bool hiHalf = (lane >= 16);

  v8f acc[2][4];
#pragma unroll
  for (int mf = 0; mf < 2; ++mf)
#pragma unroll
    for (int s = 0; s < 4; ++s) acc[mf][s] = (v8f){};

  const _Float16* arow0 = A + (size_t)(m0 + l15) * K;
  const _Float16* arow1 = arow0 + (size_t)16 * K;
  const _Float16* brow[4];
#pragma unroll
  for (int s = 0; s < 4; ++s)
    brow[s] = Bt + (size_t)(n0 + s * 16 + l15) * K;

  // Prologue: load k-step 0
  v16h aC0 = load_afrag(arow0, 0, hiHalf);
  v16h aC1 = load_afrag(arow1, 0, hiHalf);
  v16h bC0 = load_bfrag(brow[0], 0, hiHalf);
  v16h bC1 = load_bfrag(brow[1], 0, hiHalf);
  v16h bC2 = load_bfrag(brow[2], 0, hiHalf);
  v16h bC3 = load_bfrag(brow[3], 0, hiHalf);

  for (int k0 = 32; k0 < K; k0 += 32) {
    // Issue next k-step loads first (overlap with the 8 WMMAs below).
    v16h aN0 = load_afrag(arow0, k0, hiHalf);
    v16h aN1 = load_afrag(arow1, k0, hiHalf);
    v16h bN0 = load_bfrag(brow[0], k0, hiHalf);
    v16h bN1 = load_bfrag(brow[1], k0, hiHalf);
    v16h bN2 = load_bfrag(brow[2], k0, hiHalf);
    v16h bN3 = load_bfrag(brow[3], k0, hiHalf);
    if (k0 + 32 < K) {
      __builtin_prefetch(arow0 + k0 + 32, 0, 3);
      __builtin_prefetch(brow[0] + k0 + 32, 0, 3);
    }

    acc[0][0] = __builtin_amdgcn_wmma_f32_16x16x32_f16(false, aC0, false, bC0, (short)0, acc[0][0], false, false);
    acc[0][1] = __builtin_amdgcn_wmma_f32_16x16x32_f16(false, aC0, false, bC1, (short)0, acc[0][1], false, false);
    acc[0][2] = __builtin_amdgcn_wmma_f32_16x16x32_f16(false, aC0, false, bC2, (short)0, acc[0][2], false, false);
    acc[0][3] = __builtin_amdgcn_wmma_f32_16x16x32_f16(false, aC0, false, bC3, (short)0, acc[0][3], false, false);
    acc[1][0] = __builtin_amdgcn_wmma_f32_16x16x32_f16(false, aC1, false, bC0, (short)0, acc[1][0], false, false);
    acc[1][1] = __builtin_amdgcn_wmma_f32_16x16x32_f16(false, aC1, false, bC1, (short)0, acc[1][1], false, false);
    acc[1][2] = __builtin_amdgcn_wmma_f32_16x16x32_f16(false, aC1, false, bC2, (short)0, acc[1][2], false, false);
    acc[1][3] = __builtin_amdgcn_wmma_f32_16x16x32_f16(false, aC1, false, bC3, (short)0, acc[1][3], false, false);

    aC0 = aN0; aC1 = aN1;
    bC0 = bN0; bC1 = bN1; bC2 = bN2; bC3 = bN3;
  }
  // Tail k-step
  acc[0][0] = __builtin_amdgcn_wmma_f32_16x16x32_f16(false, aC0, false, bC0, (short)0, acc[0][0], false, false);
  acc[0][1] = __builtin_amdgcn_wmma_f32_16x16x32_f16(false, aC0, false, bC1, (short)0, acc[0][1], false, false);
  acc[0][2] = __builtin_amdgcn_wmma_f32_16x16x32_f16(false, aC0, false, bC2, (short)0, acc[0][2], false, false);
  acc[0][3] = __builtin_amdgcn_wmma_f32_16x16x32_f16(false, aC0, false, bC3, (short)0, acc[0][3], false, false);
  acc[1][0] = __builtin_amdgcn_wmma_f32_16x16x32_f16(false, aC1, false, bC0, (short)0, acc[1][0], false, false);
  acc[1][1] = __builtin_amdgcn_wmma_f32_16x16x32_f16(false, aC1, false, bC1, (short)0, acc[1][1], false, false);
  acc[1][2] = __builtin_amdgcn_wmma_f32_16x16x32_f16(false, aC1, false, bC2, (short)0, acc[1][2], false, false);
  acc[1][3] = __builtin_amdgcn_wmma_f32_16x16x32_f16(false, aC1, false, bC3, (short)0, acc[1][3], false, false);

  // Epilogue. C/D layout: lanes 0-15: (M=mb+i, N=col) elem i; lanes 16-31: M=mb+8+i.
#pragma unroll
  for (int mf = 0; mf < 2; ++mf) {
#pragma unroll
    for (int s = 0; s < 4; ++s) {
      const int col = n0 + s * 16 + l15;
      const float bv = bias[col];
      const int mbase = m0 + mf * 16 + (hiHalf ? 8 : 0);
#pragma unroll
      for (int i = 0; i < 8; ++i) {
        float v = acc[mf][s][i] + bv;
        v = v > 0.0f ? v : 0.0f;
        Out[(size_t)(mbase + i) * N + col] = (_Float16)v;
      }
    }
  }
}

// ---------------------------------------------------------------------------
// Heads: cls [R,21] and reg [R,84] in fp32 from f16 activations
// ---------------------------------------------------------------------------
__global__ __launch_bounds__(128) void heads_kernel(const _Float16* __restrict__ h2,
                                                    const float* __restrict__ Wc,
                                                    const float* __restrict__ bc,
                                                    const float* __restrict__ Wr,
                                                    const float* __restrict__ br,
                                                    float* __restrict__ cls,
                                                    float* __restrict__ reg) {
  const int r = blockIdx.x;
  const int j = threadIdx.x;
  if (j >= NCLS + NCLS * 4) return;
  const _Float16* x = h2 + (size_t)r * HID;
  if (j < NCLS) {
    float acc = bc[j];
    for (int k = 0; k < HID; ++k) acc += (float)x[k] * Wc[(size_t)k * NCLS + j];
    cls[r * NCLS + j] = acc;
  } else {
    const int jj = j - NCLS;
    float acc = br[jj];
    for (int k = 0; k < HID; ++k) acc += (float)x[k] * Wr[(size_t)k * (NCLS * 4) + jj];
    reg[r * (NCLS * 4) + jj] = acc;
  }
}

// ---------------------------------------------------------------------------
// Candidates: softmax + box decode + clip + validity filter
// ---------------------------------------------------------------------------
__global__ __launch_bounds__(256) void postproc_kernel(const float* __restrict__ cls,
                                                       const float* __restrict__ reg,
                                                       const float* __restrict__ props,
                                                       const int* __restrict__ imgh_p,
                                                       const int* __restrict__ imgw_p,
                                                       float* __restrict__ cand_score,
                                                       float* __restrict__ cand_box,
                                                       int* __restrict__ cand_label) {
  const int t = blockIdx.x * blockDim.x + threadIdx.x;
  if (t >= NCAND) return;
  const int r = t / NFG;
  const int klass = (t % NFG) + 1;

  const float* row = cls + r * NCLS;
  float m = row[0];
  for (int i = 1; i < NCLS; ++i) m = fmaxf(m, row[i]);
  float s = 0.0f;
  for (int i = 0; i < NCLS; ++i) s += expf(row[i] - m);
  const float score = expf(row[klass] - m) / s;

  const float px1 = props[r * 4 + 0], py1 = props[r * 4 + 1];
  const float px2 = props[r * 4 + 2], py2 = props[r * 4 + 3];
  const float w = px2 - px1, h = py2 - py1;
  const float cx = px1 + 0.5f * w, cy = py1 + 0.5f * h;
  const float* rg = reg + r * (NCLS * 4) + klass * 4;
  const float dx = rg[0], dy = rg[1];
  const float dw = fminf(rg[2], BBOX_CLIP), dh = fminf(rg[3], BBOX_CLIP);
  const float pcx = dx * w + cx, pcy = dy * h + cy;
  const float pw = expf(dw) * w, ph = expf(dh) * h;
  float bx1 = pcx - 0.5f * pw, by1 = pcy - 0.5f * ph;
  float bx2 = pcx + 0.5f * pw, by2 = pcy + 0.5f * ph;
  const float imw = (float)(*imgw_p), imh = (float)(*imgh_p);
  bx1 = fminf(fmaxf(bx1, 0.0f), imw); bx2 = fminf(fmaxf(bx2, 0.0f), imw);
  by1 = fminf(fmaxf(by1, 0.0f), imh); by2 = fminf(fmaxf(by2, 0.0f), imh);

  const bool valid = (score > SCORE_THv) && ((bx2 - bx1) >= MIN_SZ) && ((by2 - by1) >= MIN_SZ);
  cand_score[t] = valid ? score : -INFINITY;
  cand_box[t * 4 + 0] = bx1; cand_box[t * 4 + 1] = by1;
  cand_box[t * 4 + 2] = bx2; cand_box[t * 4 + 3] = by2;
  cand_label[t] = klass;
}

__device__ __forceinline__ unsigned ord_of_float(float f) {
  unsigned b = __float_as_uint(f);
  return (b & 0x80000000u) ? ~b : (b | 0x80000000u);
}
__device__ __forceinline__ float float_of_ord(unsigned u) {
  unsigned b = (u & 0x80000000u) ? (u & 0x7FFFFFFFu) : ~u;
  return __uint_as_float(b);
}

// ---------------------------------------------------------------------------
// Full 16K bitonic sort (desc by score, asc index tiebreak) in 128KB LDS.
// ---------------------------------------------------------------------------
__global__ __launch_bounds__(1024) void sort16k_kernel(const float* __restrict__ cand_score,
                                                       unsigned long long* __restrict__ keys_out) {
  extern __shared__ unsigned long long sk[];
  for (int i = threadIdx.x; i < NPAD; i += blockDim.x) {
    const float sc = (i < NCAND) ? cand_score[i] : -INFINITY;
    sk[i] = ((unsigned long long)ord_of_float(sc) << 32) |
            (unsigned long long)(0xFFFFFFFFu - (unsigned)i);
  }
  __syncthreads();
  for (int k = 2; k <= NPAD; k <<= 1) {
    for (int j = k >> 1; j > 0; j >>= 1) {
      for (int i = threadIdx.x; i < NPAD; i += blockDim.x) {
        const int ixj = i ^ j;
        if (ixj > i) {
          const unsigned long long a = sk[i], b = sk[ixj];
          if ((a < b) == ((i & k) == 0)) { sk[i] = b; sk[ixj] = a; }
        }
      }
      __syncthreads();
    }
  }
  for (int i = threadIdx.x; i < NPAD; i += blockDim.x) keys_out[i] = sk[i];
}

// ---------------------------------------------------------------------------
// Greedy NMS over top-2048 (sequential i, parallel j) + final top-100 output.
// ---------------------------------------------------------------------------
__global__ __launch_bounds__(256) void nms_out_kernel(const unsigned long long* __restrict__ keys,
                                                      const float* __restrict__ cand_box,
                                                      const int* __restrict__ cand_label,
                                                      const int* __restrict__ imgh_p,
                                                      const int* __restrict__ imgw_p,
                                                      float* __restrict__ out) {
  extern __shared__ float smem[];
  float* X1 = smem;
  float* Y1 = X1 + K2;
  float* X2 = Y1 + K2;
  float* Y2 = X2 + K2;
  float* SC = Y2 + K2;
  int* CIDX = (int*)(SC + K2);
  int* KEEP = CIDX + K2;
  unsigned long long* SKEY = (unsigned long long*)(KEEP + K2);

  const float offmul = (float)max(*imgh_p, *imgw_p) + 1.0f;

  for (int i = threadIdx.x; i < K2; i += blockDim.x) {
    const unsigned long long key = keys[i];
    float sc = float_of_ord((unsigned)(key >> 32));
    unsigned cidx = 0xFFFFFFFFu - (unsigned)(key & 0xFFFFFFFFu);
    if (cidx >= (unsigned)NCAND) { cidx = 0; sc = -INFINITY; }
    CIDX[i] = (int)cidx;
    SC[i] = sc;
    const float off = (float)cand_label[cidx] * offmul;
    X1[i] = cand_box[cidx * 4 + 0] + off;
    Y1[i] = cand_box[cidx * 4 + 1] + off;
    X2[i] = cand_box[cidx * 4 + 2] + off;
    Y2[i] = cand_box[cidx * 4 + 3] + off;
    KEEP[i] = (sc > SCORE_THv) ? 1 : 0;
  }
  __syncthreads();

  for (int i = 0; i < K2 - 1; ++i) {
    if (KEEP[i]) {
      const float ax1 = X1[i], ay1 = Y1[i], ax2 = X2[i], ay2 = Y2[i];
      const float aa = (ax2 - ax1) * (ay2 - ay1);
      for (int j = i + 1 + threadIdx.x; j < K2; j += blockDim.x) {
        if (KEEP[j]) {
          const float ix1 = fmaxf(ax1, X1[j]);
          const float iy1 = fmaxf(ay1, Y1[j]);
          const float ix2 = fminf(ax2, X2[j]);
          const float iy2 = fminf(ay2, Y2[j]);
          const float inter = fmaxf(ix2 - ix1, 0.0f) * fmaxf(iy2 - iy1, 0.0f);
          const float bb = (X2[j] - X1[j]) * (Y2[j] - Y1[j]);
          const float iou = inter / fmaxf(aa + bb - inter, 1e-9f);
          if (iou > NMS_THv) KEEP[j] = 0;
        }
      }
    }
    __syncthreads();
  }

  // Final top-100: bitonic sort of 2048 keys (desc score among kept, asc index).
  for (int i = threadIdx.x; i < K2; i += blockDim.x) {
    const float sc = KEEP[i] ? SC[i] : -INFINITY;
    SKEY[i] = ((unsigned long long)ord_of_float(sc) << 32) |
              (unsigned long long)(0xFFFFFFFFu - (unsigned)i);
  }
  __syncthreads();
  for (int k = 2; k <= K2; k <<= 1) {
    for (int j = k >> 1; j > 0; j >>= 1) {
      for (int i = threadIdx.x; i < K2; i += blockDim.x) {
        const int ixj = i ^ j;
        if (ixj > i) {
          const unsigned long long a = SKEY[i], b = SKEY[ixj];
          if ((a < b) == ((i & k) == 0)) { SKEY[i] = b; SKEY[ixj] = a; }
        }
      }
      __syncthreads();
    }
  }

  for (int o = threadIdx.x; o < NOUT; o += blockDim.x) {
    const unsigned long long key = SKEY[o];
    const float sc = float_of_ord((unsigned)(key >> 32));
    const unsigned slot = 0xFFFFFFFFu - (unsigned)(key & 0xFFFFFFFFu);
    const bool ovalid = sc > SCORE_THv;
    const int cidx = CIDX[slot];
    out[o * 4 + 0] = ovalid ? cand_box[cidx * 4 + 0] : 0.0f;
    out[o * 4 + 1] = ovalid ? cand_box[cidx * 4 + 1] : 0.0f;
    out[o * 4 + 2] = ovalid ? cand_box[cidx * 4 + 2] : 0.0f;
    out[o * 4 + 3] = ovalid ? cand_box[cidx * 4 + 3] : 0.0f;
    out[4 * NOUT + o] = ovalid ? sc : 0.0f;
    out[5 * NOUT + o] = ovalid ? (float)cand_label[cidx] : 0.0f;
  }
}

// ---------------------------------------------------------------------------
extern "C" void kernel_launch(void* const* d_in, const int* in_sizes, int n_in,
                              void* d_out, int out_size, void* d_ws, size_t ws_size,
                              hipStream_t stream) {
  const float* features  = (const float*)d_in[0];
  const float* proposals = (const float*)d_in[1];
  const float* W1 = (const float*)d_in[2];
  const float* b1 = (const float*)d_in[3];
  const float* W2 = (const float*)d_in[4];
  const float* b2 = (const float*)d_in[5];
  const float* Wc = (const float*)d_in[6];
  const float* bc = (const float*)d_in[7];
  const float* Wr = (const float*)d_in[8];
  const float* br = (const float*)d_in[9];
  const int* imgh = (const int*)d_in[10];
  const int* imgw = (const int*)d_in[11];
  float* out = (float*)d_out;

  char* wp = (char*)d_ws;
  auto walloc = [&](size_t bytes) -> void* {
    void* p = (void*)wp;
    wp += (bytes + 255) & ~(size_t)255;
    return p;
  };
  _Float16* poolA = (_Float16*)walloc((size_t)RN * DN * sizeof(_Float16));
  _Float16* W1T   = (_Float16*)walloc((size_t)DN * HID * sizeof(_Float16));
  _Float16* W2T   = (_Float16*)walloc((size_t)HID * HID * sizeof(_Float16));
  _Float16* h1    = (_Float16*)walloc((size_t)RN * HID * sizeof(_Float16));
  _Float16* h2    = (_Float16*)walloc((size_t)RN * HID * sizeof(_Float16));
  float* clsS     = (float*)walloc((size_t)RN * NCLS * sizeof(float));
  float* regS     = (float*)walloc((size_t)RN * NCLS * 4 * sizeof(float));
  float* cand_score = (float*)walloc((size_t)NCAND * sizeof(float));
  float* cand_box   = (float*)walloc((size_t)NCAND * 4 * sizeof(float));
  int*   cand_label = (int*)walloc((size_t)NCAND * sizeof(int));
  unsigned long long* keys = (unsigned long long*)walloc((size_t)NPAD * 8);

  // Weight conversion (W1/W2 are L2-resident after first touch: 192MB L2)
  convt_kernel<<<dim3(DN / 32, HID / 32), dim3(32, 8), 0, stream>>>(W1, W1T, DN, HID);
  convt_kernel<<<dim3(HID / 32, HID / 32), dim3(32, 8), 0, stream>>>(W2, W2T, HID, HID);

  roi_pool_kernel<<<RN, 256, 0, stream>>>(features, proposals, imgw, poolA);

  gemm_wmma_kernel<<<dim3(RN / 64, HID / 128), 128, 0, stream>>>(poolA, W1T, b1, h1, RN, HID, DN);
  gemm_wmma_kernel<<<dim3(RN / 64, HID / 128), 128, 0, stream>>>(h1, W2T, b2, h2, RN, HID, HID);

  heads_kernel<<<RN, 128, 0, stream>>>(h2, Wc, bc, Wr, br, clsS, regS);

  postproc_kernel<<<(NCAND + 255) / 256, 256, 0, stream>>>(clsS, regS, proposals, imgh, imgw,
                                                           cand_score, cand_box, cand_label);

  sort16k_kernel<<<1, 1024, NPAD * sizeof(unsigned long long), stream>>>(cand_score, keys);

  const size_t nms_lds = (size_t)K2 * (5 * 4 + 2 * 4 + 8);  // 73728 B (< 320KB WGP LDS)
  nms_out_kernel<<<1, 256, nms_lds, stream>>>(keys, cand_box, cand_label, imgh, imgw, out);
}
